// SumThenRegRpeSelfAttention_7103875908187
// MI455X (gfx1250) — compile-verified, matrix operand order checked by reading, not run on hardware
//
#include <hip/hip_runtime.h>
#include <hip/hip_bf16.h>

// ---------------- problem constants ----------------
#define Bv   4
#define Tv   1024
#define Ev   768
#define Hv   12
#define DHv  64
#define Pv   65            // MAXP + 1
#define SCALEv 0.125f      // DH^-0.5 = 1/8

typedef __attribute__((ext_vector_type(16))) __bf16 v16bf;
typedef __attribute__((ext_vector_type(8)))  __bf16 v8bf;
typedef __attribute__((ext_vector_type(8)))  float  v8f;

// ---------------- WMMA helpers ----------------
// 16-bit A-fragment (16x32, row-major source, row stride ld in elements).
// Per ISA layout: lanes 0-15 -> rows, chunk0 = K[sel*8 .. sel*8+7],
// chunk1 = K[16+sel*8 .. 16+sel*8+7], sel = lane>>4.
__device__ __forceinline__ v16bf load_frag_a(const __bf16* p, int ld) {
  const int lane = threadIdx.x & 31;
  const __bf16* r = p + (size_t)(lane & 15) * ld + ((lane >> 4) << 3);
  v8bf lo = *reinterpret_cast<const v8bf*>(r);
  v8bf hi = *reinterpret_cast<const v8bf*>(r + 16);
  v16bf f;
#pragma unroll
  for (int i = 0; i < 8; ++i) { f[i] = lo[i]; f[i + 8] = hi[i]; }
  return f;
}

__device__ __forceinline__ v8f wmma_bf16(v16bf a, v16bf b, v8f c) {
  return __builtin_amdgcn_wmma_f32_16x16x32_bf16(false, a, false, b,
                                                 (short)0, c, false, false);
}

// ---------------- elementwise / small kernels ----------------
__global__ void cvt_bf16(const float* __restrict__ x, __bf16* __restrict__ y, int n) {
  int i = blockIdx.x * blockDim.x + threadIdx.x;
  if (i < n) y[i] = (__bf16)x[i];
}

// table[p,n] = sum_k emb[p,k] * Wr[n,k] + br[n]   (65x768, f32, tiny)
__global__ void table_kernel(const float* __restrict__ emb, const float* __restrict__ Wr,
                             const float* __restrict__ br, float* __restrict__ tbl) {
  int idx = blockIdx.x * blockDim.x + threadIdx.x;
  if (idx >= Pv * Ev) return;
  int p = idx / Ev, n = idx % Ev;
  const float* e = emb + (size_t)p * Ev;
  const float* w = Wr + (size_t)n * Ev;
  float acc = br[n];
  for (int k = 0; k < Ev; ++k) acc += e[k] * w[k];
  tbl[idx] = acc;
}

// ts[bh,q,p] = sum_d q[b,q,h*DH+d] * table[p, h*DH+d]
__global__ void ts_kernel(const __bf16* __restrict__ Q, const float* __restrict__ tbl,
                          float* __restrict__ TS) {
  size_t idx = (size_t)blockIdx.x * blockDim.x + threadIdx.x;
  const size_t total = (size_t)Bv * Hv * Tv * Pv;
  if (idx >= total) return;
  int p = (int)(idx % Pv);
  size_t qi = idx / Pv;          // bh*T + q
  int q  = (int)(qi % Tv);
  int bh = (int)(qi / Tv);
  int b = bh / Hv, h = bh % Hv;
  const __bf16* qp = Q + ((size_t)b * Tv + q) * Ev + h * DHv;
  const float*  tp = tbl + (size_t)p * Ev + h * DHv;
  float acc = 0.f;
#pragma unroll 8
  for (int d = 0; d < DHv; ++d) acc += (float)qp[d] * tp[d];
  TS[idx] = acc;
}

// [B,T,E] head slice -> [B,H,DH,T]
__global__ void transpose_head(const __bf16* __restrict__ X, __bf16* __restrict__ XT) {
  size_t idx = (size_t)blockIdx.x * blockDim.x + threadIdx.x;
  const size_t total = (size_t)Bv * Hv * DHv * Tv;
  if (idx >= total) return;
  int t = (int)(idx % Tv);
  size_t r = idx / Tv;
  int d  = (int)(r % DHv);
  int bh = (int)(r / DHv);
  int b = bh / Hv, h = bh % Hv;
  XT[idx] = X[((size_t)b * Tv + t) * Ev + h * DHv + d];
}

// ---------------- WMMA GEMM: Y = alpha*(X @ W^T + bias) ----------------
// X: [M, E] bf16 row-major, W: [E, E] bf16 row-major (rows = output features).
// Wave computes a 32(M) x 64(N) tile; block = 4 waves stacked along M, all
// sharing the same 64-row B panel. The B panel is staged into LDS with the
// CDNA5 async global->LDS path (ASYNCcnt), double-buffered.
__global__ __launch_bounds__(128)
void gemm_bias(const __bf16* __restrict__ X, const __bf16* __restrict__ W,
               const float* __restrict__ bias, void* __restrict__ Y,
               int M, float alpha, int out_f32) {
  __shared__ __bf16 bbuf[2][64 * 32];   // two 4KB B panels (64 rows x 32 bf16)

  const int lane = threadIdx.x & 31;
  const int wave = threadIdx.x >> 5;
  const int m0 = blockIdx.x * 128 + wave * 32;
  const int n0 = blockIdx.y * 64;

  // Cooperative async stage of B panel for K-step `it` into bbuf[it&1].
  // Thread t moves 32 bytes of row (t>>1): two async b128 ops; INST_OFFSET
  // is added to both the global and LDS addresses (ISA §10, async pseudocode).
  auto stage = [&](int it) {
    __bf16* buf = bbuf[it & 1];
    const int t   = threadIdx.x;
    const int row = t >> 1;
    const int cb  = (t & 1) * 16;                       // element offset in row
    const __bf16* g = W + (size_t)(n0 + row) * Ev + it * 32 + cb;
    unsigned laddr = (unsigned)(size_t)(const void*)(buf + row * 32 + cb);
    unsigned long long ga = (unsigned long long)(size_t)g;
    asm volatile("global_load_async_to_lds_b128 %0, %1, off"
                 :: "v"(laddr), "v"(ga) : "memory");
    asm volatile("global_load_async_to_lds_b128 %0, %1, off offset:16"
                 :: "v"(laddr), "v"(ga) : "memory");
  };

  v8f acc[2][4];
#pragma unroll
  for (int i = 0; i < 2; ++i)
#pragma unroll
    for (int j = 0; j < 4; ++j) acc[i][j] = (v8f){};

  stage(0);
  const int NIT = Ev / 32;                 // 24 K-steps
  for (int it = 0; it < NIT; ++it) {
    asm volatile("s_wait_asynccnt 0x0" ::: "memory");
    __syncthreads();                       // bbuf[it&1] visible to all waves
    if (it + 1 < NIT) stage(it + 1);       // prefetch next panel into other buf

    const __bf16* cur = bbuf[it & 1];
    const int kk = it * 32;
    v16bf a0 = load_frag_a(X + (size_t)m0 * Ev + kk, Ev);
    v16bf a1 = load_frag_a(X + (size_t)(m0 + 16) * Ev + kk, Ev);
#pragma unroll
    for (int j = 0; j < 4; ++j) {
      v16bf bm = load_frag_a(cur + j * 16 * 32, 32);   // ds_load_b128 x2
      acc[0][j] = wmma_bf16(a0, bm, acc[0][j]);
      acc[1][j] = wmma_bf16(a1, bm, acc[1][j]);
    }
  }

  const int half = (lane >> 4) * 8;
  const int col  = lane & 15;
#pragma unroll
  for (int i = 0; i < 2; ++i)
#pragma unroll
    for (int j = 0; j < 4; ++j) {
      int n = n0 + j * 16 + col;
      float bn = bias[n];
#pragma unroll
      for (int r = 0; r < 8; ++r) {
        int m = m0 + i * 16 + half + r;
        float v = alpha * (acc[i][j][r] + bn);
        if (out_f32) ((float*)Y)[(size_t)m * Ev + n] = v;
        else         ((__bf16*)Y)[(size_t)m * Ev + n] = (__bf16)v;
      }
    }
  (void)M;
}

// ---------------- Flash attention (shared by both branches) ----------------
// Q,K,K2 : [B,T,E] bf16 head-sliced; VT,V2T : [B,H,DH,T] bf16.
// If K2 != null: second key set (summary keys) continues the online softmax.
// If TS0 != null: RPE bias ts0[bh,q,rp0[q,k]] + ts1[bh,q,rp1[q,k]] added to
// first-set scores. Wave handles 16 queries; key blocks of 32.
__global__ __launch_bounds__(128)
void flash_attn(const __bf16* __restrict__ Q, const __bf16* __restrict__ K,
                const __bf16* __restrict__ VT,
                const __bf16* __restrict__ K2, const __bf16* __restrict__ V2T,
                const float* __restrict__ TS0, const float* __restrict__ TS1,
                const int* __restrict__ RP0, const int* __restrict__ RP1,
                __bf16* __restrict__ OUT) {
  __shared__ __bf16 pbuf[4][16 * 32];

  const int lane = threadIdx.x & 31;
  const int wave = threadIdx.x >> 5;
  const int gw   = blockIdx.x * 4 + wave;     // B*H*(T/16) = 3072 waves
  const int qblk = gw & 63;                   // T/16
  const int bh   = gw >> 6;                   // 0..47
  const int b = bh / Hv, h = bh % Hv;
  const int q0 = qblk * 16;
  const int half = (lane >> 4) * 8;
  const int col  = lane & 15;
  __bf16* pl = pbuf[wave];

  const __bf16* qbase = Q + ((size_t)b * Tv) * Ev + h * DHv;
  v16bf aQ0 = load_frag_a(qbase + (size_t)q0 * Ev, Ev);
  v16bf aQ1 = load_frag_a(qbase + (size_t)q0 * Ev + 32, Ev);

  v8f o[4];
#pragma unroll
  for (int t = 0; t < 4; ++t) o[t] = (v8f){};
  float mrun[8], lrun[8];
#pragma unroll
  for (int r = 0; r < 8; ++r) { mrun[r] = -1e30f; lrun[r] = 0.f; }

  for (int s = 0; s < 2; ++s) {
    const __bf16* kbase;
    const __bf16* vtb;
    if (s == 0) {
      kbase = K + ((size_t)b * Tv) * Ev + h * DHv;
      vtb   = VT + ((size_t)bh * DHv) * Tv;
    } else {
      if (!K2) break;
      kbase = K2 + ((size_t)b * Tv) * Ev + h * DHv;
      vtb   = V2T + ((size_t)bh * DHv) * Tv;
    }
    const bool use_bias = (s == 0) && (TS0 != nullptr);

    for (int kb = 0; kb < Tv; kb += 32) {
      const __bf16* kb0 = kbase + (size_t)kb * Ev;
      const __bf16* kb1 = kbase + (size_t)(kb + 16) * Ev;
      if (kb + 32 < Tv)
        __builtin_prefetch(kbase + (size_t)(kb + 32) * Ev, 0, 1);  // global_prefetch_b8
      v8f c0 = (v8f){}, c1 = (v8f){};
      c0 = wmma_bf16(aQ0, load_frag_a(kb0, Ev), c0);
      c0 = wmma_bf16(aQ1, load_frag_a(kb0 + 32, Ev), c0);
      c1 = wmma_bf16(aQ0, load_frag_a(kb1, Ev), c1);
      c1 = wmma_bf16(aQ1, load_frag_a(kb1 + 32, Ev), c1);

      if (use_bias) {
#pragma unroll
        for (int r = 0; r < 8; ++r) {
          int qg = q0 + half + r;
          const float* t0 = TS0 + ((size_t)bh * Tv + qg) * Pv;
          const float* t1 = TS1 + ((size_t)bh * Tv + qg) * Pv;
          int kg0 = kb + col, kg1 = kb + 16 + col;
          c0[r] += t0[RP0[(size_t)qg * Tv + kg0]] + t1[RP1[(size_t)qg * Tv + kg0]];
          c1[r] += t0[RP0[(size_t)qg * Tv + kg1]] + t1[RP1[(size_t)qg * Tv + kg1]];
        }
      }

      // online softmax per row (row stats live in 16-lane halves)
#pragma unroll
      for (int r = 0; r < 8; ++r) {
        float mx = fmaxf(c0[r], c1[r]);
        for (int off = 1; off < 16; off <<= 1) mx = fmaxf(mx, __shfl_xor(mx, off, 32));
        float mnew  = fmaxf(mrun[r], mx);
        float scale = __expf(mrun[r] - mnew);
        float p0 = __expf(c0[r] - mnew);
        float p1 = __expf(c1[r] - mnew);
        float rs = p0 + p1;
        for (int off = 1; off < 16; off <<= 1) rs += __shfl_xor(rs, off, 32);
        lrun[r] = lrun[r] * scale + rs;
        mrun[r] = mnew;
#pragma unroll
        for (int t = 0; t < 4; ++t) o[t][r] *= scale;
        c0[r] = p0; c1[r] = p1;
      }

      // stage P (16x32) through LDS to re-layout C-tile -> A-fragment
#pragma unroll
      for (int r = 0; r < 8; ++r) {
        pl[(half + r) * 32 + col]      = (__bf16)c0[r];
        pl[(half + r) * 32 + 16 + col] = (__bf16)c1[r];
      }
      asm volatile("s_wait_dscnt 0" ::: "memory");
      v16bf aP = load_frag_a(pl, 32);

#pragma unroll
      for (int t = 0; t < 4; ++t) {
        v16bf bv = load_frag_a(vtb + (size_t)(t * 16) * Tv + kb, Tv);
        o[t] = wmma_bf16(aP, bv, o[t]);
      }
    }
  }

  // epilogue: normalize and store bf16 into [B,T,E] head slice
#pragma unroll
  for (int r = 0; r < 8; ++r) {
    float inv = 1.0f / lrun[r];
    int m = q0 + half + r;
    __bf16* orow = OUT + ((size_t)b * Tv + m) * Ev + h * DHv;
#pragma unroll
    for (int t = 0; t < 4; ++t) orow[t * 16 + col] = (__bf16)(o[t][r] * inv);
  }
}

// ---------------- host orchestration ----------------
extern "C" void kernel_launch(void* const* d_in, const int* in_sizes, int n_in,
                              void* d_out, int out_size, void* d_ws, size_t ws_size,
                              hipStream_t stream) {
  (void)in_sizes; (void)n_in; (void)out_size; (void)ws_size;

  const float* x     = (const float*)d_in[0];
  const int*   rp0   = (const int*)d_in[1];
  const int*   rp1   = (const int*)d_in[2];
  const float* emb0  = (const float*)d_in[3];
  const float* emb1  = (const float*)d_in[4];
  const float* Wq_r  = (const float*)d_in[5];   const float* bq_r = (const float*)d_in[6];
  const float* Wk_r  = (const float*)d_in[7];   const float* bk_r = (const float*)d_in[8];
  const float* Wv_r  = (const float*)d_in[9];   const float* bv_r = (const float*)d_in[10];
  const float* Wq_s  = (const float*)d_in[11];  const float* bq_s = (const float*)d_in[12];
  const float* Wk_s  = (const float*)d_in[13];  const float* bk_s = (const float*)d_in[14];
  const float* Wv_s  = (const float*)d_in[15];  const float* bv_s = (const float*)d_in[16];
  const float* Wk2   = (const float*)d_in[17];  const float* bk2  = (const float*)d_in[18];
  const float* Wv2   = (const float*)d_in[19];  const float* bv2  = (const float*)d_in[20];
  const float* Wrel0 = (const float*)d_in[21];  const float* brel0= (const float*)d_in[22];
  const float* Wrel1 = (const float*)d_in[23];  const float* brel1= (const float*)d_in[24];
  const float* Wo    = (const float*)d_in[25];  const float* bo   = (const float*)d_in[26];

  const size_t NTOK = (size_t)Bv * Tv;           // 4096
  const size_t ACT  = NTOK * Ev;                 // 3,145,728 elems
  const size_t WSZ  = (size_t)Ev * Ev;           // 589,824 elems

  char* ws = (char*)d_ws;
  size_t off = 0;
  auto alloc = [&](size_t bytes) -> void* {
    void* p = ws + off;
    off = (off + bytes + 255) & ~(size_t)255;
    return p;
  };

  __bf16* xb  = (__bf16*)alloc(ACT * 2);
  __bf16* wqr = (__bf16*)alloc(WSZ * 2);
  __bf16* wkr = (__bf16*)alloc(WSZ * 2);
  __bf16* wvr = (__bf16*)alloc(WSZ * 2);
  __bf16* wqs = (__bf16*)alloc(WSZ * 2);
  __bf16* wks = (__bf16*)alloc(WSZ * 2);
  __bf16* wvs = (__bf16*)alloc(WSZ * 2);
  __bf16* wk2 = (__bf16*)alloc(WSZ * 2);
  __bf16* wv2 = (__bf16*)alloc(WSZ * 2);
  __bf16* wob = (__bf16*)alloc(WSZ * 2);
  __bf16* qsb  = (__bf16*)alloc(ACT * 2);
  __bf16* ksb  = (__bf16*)alloc(ACT * 2);
  __bf16* vsb  = (__bf16*)alloc(ACT * 2);
  __bf16* qb   = (__bf16*)alloc(ACT * 2);
  __bf16* kb   = (__bf16*)alloc(ACT * 2);
  __bf16* vb   = (__bf16*)alloc(ACT * 2);
  __bf16* sumx = (__bf16*)alloc(ACT * 2);
  __bf16* k2b  = (__bf16*)alloc(ACT * 2);
  __bf16* v2b  = (__bf16*)alloc(ACT * 2);
  __bf16* outx = (__bf16*)alloc(ACT * 2);
  __bf16* vst  = (__bf16*)alloc(ACT * 2);
  __bf16* vrt  = (__bf16*)alloc(ACT * 2);
  __bf16* v2t  = (__bf16*)alloc(ACT * 2);
  float* tbl0 = (float*)alloc((size_t)Pv * Ev * 4);
  float* tbl1 = (float*)alloc((size_t)Pv * Ev * 4);
  float* ts0  = (float*)alloc((size_t)Bv * Hv * Tv * Pv * 4);
  float* ts1  = (float*)alloc((size_t)Bv * Hv * Tv * Pv * 4);

  // 1) bf16 conversions
  cvt_bf16<<<(int)((ACT + 255) / 256), 256, 0, stream>>>(x, xb, (int)ACT);
  const float* wsrc[9] = {Wq_r, Wk_r, Wv_r, Wq_s, Wk_s, Wv_s, Wk2, Wv2, Wo};
  __bf16*      wdst[9] = {wqr,  wkr,  wvr,  wqs,  wks,  wvs,  wk2, wv2, wob};
  for (int i = 0; i < 9; ++i)
    cvt_bf16<<<(int)((WSZ + 255) / 256), 256, 0, stream>>>(wsrc[i], wdst[i], (int)WSZ);

  // 2) projections (WMMA GEMMs); SCALE folded into q projections
  dim3 gg((unsigned)(NTOK / 128), (unsigned)(Ev / 64));
  gemm_bias<<<gg, 128, 0, stream>>>(xb, wqs, bq_s, qsb, (int)NTOK, SCALEv, 0);
  gemm_bias<<<gg, 128, 0, stream>>>(xb, wks, bk_s, ksb, (int)NTOK, 1.0f, 0);
  gemm_bias<<<gg, 128, 0, stream>>>(xb, wvs, bv_s, vsb, (int)NTOK, 1.0f, 0);
  gemm_bias<<<gg, 128, 0, stream>>>(xb, wqr, bq_r, qb,  (int)NTOK, SCALEv, 0);
  gemm_bias<<<gg, 128, 0, stream>>>(xb, wkr, bk_r, kb,  (int)NTOK, 1.0f, 0);
  gemm_bias<<<gg, 128, 0, stream>>>(xb, wvr, bv_r, vb,  (int)NTOK, 1.0f, 0);

  // 3) sum branch: transpose V, flash attention -> sum_x
  const int ntr = (int)((ACT + 255) / 256);
  transpose_head<<<ntr, 256, 0, stream>>>(vsb, vst);
  flash_attn<<<768, 128, 0, stream>>>(qsb, ksb, vst, nullptr, nullptr,
                                      nullptr, nullptr, nullptr, nullptr, sumx);

  // 4) summary keys/values, transposes
  gemm_bias<<<gg, 128, 0, stream>>>(sumx, wk2, bk2, k2b, (int)NTOK, 1.0f, 0);
  gemm_bias<<<gg, 128, 0, stream>>>(sumx, wv2, bv2, v2b, (int)NTOK, 1.0f, 0);
  transpose_head<<<ntr, 256, 0, stream>>>(vb,  vrt);
  transpose_head<<<ntr, 256, 0, stream>>>(v2b, v2t);

  // 5) RPE tables + ts
  table_kernel<<<(Pv * Ev + 255) / 256, 256, 0, stream>>>(emb0, Wrel0, brel0, tbl0);
  table_kernel<<<(Pv * Ev + 255) / 256, 256, 0, stream>>>(emb1, Wrel1, brel1, tbl1);
  const size_t nts = (size_t)Bv * Hv * Tv * Pv;
  ts_kernel<<<(int)((nts + 255) / 256), 256, 0, stream>>>(qb, tbl0, ts0);
  ts_kernel<<<(int)((nts + 255) / 256), 256, 0, stream>>>(qb, tbl1, ts1);

  // 6) reg branch flash attention over 2T keys with RPE bias
  flash_attn<<<768, 128, 0, stream>>>(qb, kb, vrt, k2b, v2t,
                                      ts0, ts1, rp0, rp1, outx);

  // 7) output projection -> f32 d_out
  gemm_bias<<<gg, 128, 0, stream>>>(outx, wob, bo, d_out, (int)NTOK, 1.0f, 1);
}
